// LayoutGraphModel_71786083385992
// MI455X (gfx1250) — matrix operand order repaired and checked by reading, not compiled, not run on hardware
//
#include <hip/hip_runtime.h>
#include <hip/hip_bf16.h>
#include <cstdint>

// ---------------- WMMA types (CDNA5 / gfx1250, wave32) ----------------
typedef __attribute__((ext_vector_type(16))) _Float16 v16h;
typedef __attribute__((ext_vector_type(8)))  float    v8f;

#define WAVE 32

// ---------------- x0 = concat(node_features, emb[node_ops]) ----------------
__global__ void concat_embed_kernel(const float* __restrict__ nf,
                                    const int* __restrict__ ops,
                                    const float* __restrict__ emb,
                                    float* __restrict__ x,
                                    _Float16* __restrict__ xh,
                                    int N) {
  int idx = blockIdx.x * blockDim.x + threadIdx.x;
  if (idx >= N * 128) return;
  int n = idx >> 7, c = idx & 127;
  float v = (c < 64) ? nf[n * 64 + c] : emb[ops[n] * 64 + (c - 64)];
  x[idx]  = v;
  xh[idx] = (_Float16)v;
}

// ---------------- degree / inverse degree ----------------
__global__ void degree_kernel(const int* __restrict__ dst, float* __restrict__ deg, int E) {
  int e = blockIdx.x * blockDim.x + threadIdx.x;
  if (e < E) atomicAdd(&deg[dst[e]], 1.0f);
}

__global__ void invdeg_kernel(const float* __restrict__ deg, float* __restrict__ inv, int N) {
  int n = blockIdx.x * blockDim.x + threadIdx.x;
  if (n < N) inv[n] = 1.0f / fmaxf(deg[n], 1.0f);
}

// ---------------- scatter-add: agg[dst] += x[src] (float4 chunks) ----------------
__global__ void scatter_add_kernel(const float* __restrict__ x,
                                   const int* __restrict__ src,
                                   const int* __restrict__ dst,
                                   float* __restrict__ agg,
                                   int E, int d) {
  int chunks = d >> 2;
  long long idx = (long long)blockIdx.x * blockDim.x + threadIdx.x;
  if (idx >= (long long)E * chunks) return;
  int e = (int)(idx / chunks);
  int c = (int)(idx % chunks);
  int s = src[e], t = dst[e];
  const float* xrow = x + (size_t)s * d;
  float* orow = agg + (size_t)t * d + (size_t)c * 4;
  __builtin_prefetch(orow, 1, 0);            // global_prefetch of the scatter target line
  float4 v = reinterpret_cast<const float4*>(xrow)[c];
  atomicAdd(orow + 0, v.x);
  atomicAdd(orow + 1, v.y);
  atomicAdd(orow + 2, v.z);
  atomicAdd(orow + 3, v.w);
}

// ---------------- mean-scale + f32 -> f16 convert ----------------
__global__ void scale_convert_kernel(const float* __restrict__ agg,
                                     const float* __restrict__ invdeg,
                                     _Float16* __restrict__ aggh,
                                     int total, int shift) {
  int idx = blockIdx.x * blockDim.x + threadIdx.x;
  if (idx >= total) return;
  int n = idx >> shift;
  aggh[idx] = (_Float16)(agg[idx] * invdeg[n]);
}

// ---------------- repack weights [K,N] f32 -> WMMA B fragment order, f16 ----------------
// B frag element e[j] of lane l for tile (t, s): K = s*32 + (l>>4)*16 + j, N = t*16 + (l&15)
// Packed layout: Wpk[(((t*(K/32)+s)*32)+l)*16 + j]  (contiguous 32B per lane per k-step)
__global__ void repack_w_kernel(const float* __restrict__ W,
                                _Float16* __restrict__ Wpk,
                                int K, int N) {
  int idx = blockIdx.x * blockDim.x + threadIdx.x;
  if (idx >= K * N) return;
  int j    = idx & 15;
  int lane = (idx >> 4) & 31;
  int rest = idx >> 9;
  int ks   = K >> 5;
  int s    = rest % ks;
  int t    = rest / ks;
  int srcK = s * 32 + ((lane >> 4) << 4) + j;
  int srcN = t * 16 + (lane & 15);
  Wpk[idx] = (_Float16)W[(size_t)srcK * N + srcN];
}

// ---------------- WMMA K-loop over 4 N-tiles: c[t] += A * Wpk[:, tN0+t] ----------------
// A: row-major [M, K] f16, lda == K (K multiple of 32).
// A fragment as loaded here (contiguous k, same order as Wpk): lanes 0-15 -> K=0..15; lanes 16-31 -> K=16..31.
__device__ __forceinline__ void wmma_accum4(const _Float16* __restrict__ A,
                                            const _Float16* __restrict__ Wpk,
                                            int K, int tM, int tN0, int lane,
                                            v8f (&c)[4]) {
  union Frag { v16h v; uint4 u4[2]; };
  const int row   = tM * 16 + (lane & 15);
  const int khalf = (lane >> 4) * 16;
  const int ks    = K >> 5;
  const uint4* ap = reinterpret_cast<const uint4*>(A + (size_t)row * K + khalf);
  const uint4* bp = reinterpret_cast<const uint4*>(Wpk + ((size_t)tN0 * ks * 32 + lane) * 16);
  const size_t btile = (size_t)ks * 64;   // uint4 stride between adjacent N-tiles
  for (int s = 0; s < ks; ++s) {
    Frag a;
    a.u4[0] = ap[0];        // K = k + khalf .. +7
    a.u4[1] = ap[1];        // K = k + khalf + 8 .. +15
    ap += 4;                // advance 32 halfs
    #pragma unroll
    for (int t = 0; t < 4; ++t) {
      Frag b;
      b.u4[0] = bp[(size_t)t * btile + 0];
      b.u4[1] = bp[(size_t)t * btile + 1];
      c[t] = __builtin_amdgcn_wmma_f32_16x16x32_f16(
          /*neg_a=*/false, a.v, /*neg_b=*/false, b.v,
          /*c_mod=*/(short)0, c[t], /*reuse_a=*/false, /*reuse_b=*/false);
    }
    bp += 64;               // next k-step within each tile
  }
}

// ---------------- out = A1 @ W1 + bias + A2 @ W2  (one wave per 16x64 tile) ----------------
__global__ void gemm_dual_kernel(const _Float16* __restrict__ A1,
                                 const _Float16* __restrict__ W1,
                                 const _Float16* __restrict__ A2,
                                 const _Float16* __restrict__ W2,
                                 const float* __restrict__ bias,
                                 float* __restrict__ out,
                                 int M, int K, int dO) {
  const int lane = threadIdx.x & (WAVE - 1);
  const int wid  = blockIdx.x * (blockDim.x / WAVE) + (threadIdx.x / WAVE);
  const int ng   = dO >> 6;           // groups of 4 N-tiles
  const int tM   = wid / ng;
  const int tN0  = (wid % ng) * 4;
  if (tM * 16 >= M) return;           // wave-uniform

  v8f c[4] = {{}, {}, {}, {}};
  wmma_accum4(A1, W1, K, tM, tN0, lane, c);
  wmma_accum4(A2, W2, K, tM, tN0, lane, c);

  const int row0 = tM * 16 + ((lane >> 4) << 3);
  #pragma unroll
  for (int t = 0; t < 4; ++t) {
    const int col = (tN0 + t) * 16 + (lane & 15);
    const float bv = bias[col];
    #pragma unroll
    for (int r = 0; r < 8; ++r)
      out[(size_t)(row0 + r) * dO + col] = c[t][r] + bv;
  }
}

// ---------------- row-wise L2 normalize + LeakyReLU (wave32 per row) ----------------
__global__ void norm_act_kernel(const float* __restrict__ in,
                                float* __restrict__ out,
                                _Float16* __restrict__ outh,
                                int N, int d) {
  const int lane = threadIdx.x & (WAVE - 1);
  const int row  = blockIdx.x * (blockDim.x / WAVE) + (threadIdx.x / WAVE);
  if (row >= N) return;
  const float* r = in + (size_t)row * d;
  float s = 0.f;
  for (int j = lane; j < d; j += WAVE) { float v = r[j]; s += v * v; }
  #pragma unroll
  for (int m = 16; m >= 1; m >>= 1) s += __shfl_xor(s, m, WAVE);
  const float inv = 1.0f / fmaxf(sqrtf(s), 1e-12f);
  for (int j = lane; j < d; j += WAVE) {
    float v = r[j] * inv;
    v = (v > 0.f) ? v : 0.01f * v;
    out[(size_t)row * d + j]  = v;
    outh[(size_t)row * d + j] = (_Float16)v;
  }
}

// ---------------- orchestration ----------------
extern "C" void kernel_launch(void* const* d_in, const int* in_sizes, int n_in,
                              void* d_out, int out_size, void* d_ws, size_t ws_size,
                              hipStream_t stream) {
  const float* nf   = (const float*)d_in[0];
  const int*   ops  = (const int*)d_in[1];
  const int*   edges= (const int*)d_in[2];
  // d_in[3] node_separation, d_in[4] batches: unused by the reference output
  const float* emb  = (const float*)d_in[5];
  const float* Wl1  = (const float*)d_in[6];
  const float* bl1  = (const float*)d_in[7];
  const float* Wr1  = (const float*)d_in[8];
  const float* Wl2  = (const float*)d_in[9];
  const float* bl2  = (const float*)d_in[10];
  const float* Wr2  = (const float*)d_in[11];
  const float* Wl3  = (const float*)d_in[12];
  const float* bl3  = (const float*)d_in[13];
  const float* Wr3  = (const float*)d_in[14];

  const int N = in_sizes[1];
  const int E = in_sizes[2] / 2;
  const int* src = edges;
  const int* dst = edges + E;
  const int H = 256;

  // workspace carve-out
  char* ws = (char*)d_ws;
  size_t off = 0;
  auto carve = [&](size_t bytes) -> void* {
    void* p = ws + off;
    off = (off + bytes + 255) & ~(size_t)255;
    return p;
  };
  float*    bufX  = (float*)   carve((size_t)N * H * 4);
  _Float16* bufXh = (_Float16*)carve((size_t)N * H * 2);
  float*    agg   = (float*)   carve((size_t)N * H * 4);   // also reused as GEMM output
  _Float16* aggh  = (_Float16*)carve((size_t)N * H * 2);
  float*    deg   = (float*)   carve((size_t)N * 4);
  float*    invd  = (float*)   carve((size_t)N * 4);
  const float* wsrc[6] = {Wl1, Wr1, Wl2, Wr2, Wl3, Wr3};
  const int    wk[6]   = {128, 128, 256, 256, 256, 256};
  _Float16* wpk[6];
  for (int i = 0; i < 6; ++i) wpk[i] = (_Float16*)carve((size_t)wk[i] * H * 2);

  const int TB = 256;

  // x0 = concat(features, emb[ops]) (f32 + f16)
  concat_embed_kernel<<<(N * 128 + TB - 1) / TB, TB, 0, stream>>>(nf, ops, emb, bufX, bufXh, N);

  // degrees (shared across layers)
  hipMemsetAsync(deg, 0, (size_t)N * 4, stream);
  degree_kernel<<<(E + TB - 1) / TB, TB, 0, stream>>>(dst, deg, E);
  invdeg_kernel<<<(N + TB - 1) / TB, TB, 0, stream>>>(deg, invd, N);

  // repack all weights to WMMA B-fragment order (f16)
  for (int i = 0; i < 6; ++i) {
    int elems = wk[i] * H;
    repack_w_kernel<<<(elems + TB - 1) / TB, TB, 0, stream>>>(wsrc[i], wpk[i], wk[i], H);
  }

  const int   din[3]  = {128, 256, 256};
  const float* bls[3] = {bl1, bl2, bl3};

  for (int L = 0; L < 3; ++L) {
    const int K = din[L];
    const int shift = (K == 128) ? 7 : 8;

    // mean aggregation of neighbor features
    hipMemsetAsync(agg, 0, (size_t)N * K * 4, stream);
    long long work = (long long)E * (K >> 2);
    scatter_add_kernel<<<(unsigned)((work + TB - 1) / TB), TB, 0, stream>>>(bufX, src, dst, agg, E, K);
    scale_convert_kernel<<<(N * K + TB - 1) / TB, TB, 0, stream>>>(agg, invd, aggh, N * K, shift);

    // out = agg @ Wl + bl + x @ Wr   (WMMA f16 -> f32 accum, 16x64 per wave)
    float* outp = (L == 2) ? (float*)d_out : agg;  // agg f32 is free after conversion
    int tilesM = (N + 15) / 16;
    int waves  = tilesM * (H / 64);
    int wavesPerBlock = TB / WAVE;
    gemm_dual_kernel<<<(waves + wavesPerBlock - 1) / wavesPerBlock, TB, 0, stream>>>(
        aggh, wpk[2 * L], bufXh, wpk[2 * L + 1], bls[L], outp, N, K, H);

    // L2 normalize + LeakyReLU, feed next layer (f32 + f16)
    if (L < 2) {
      int rowsPerBlock = TB / WAVE;
      norm_act_kernel<<<(N + rowsPerBlock - 1) / rowsPerBlock, TB, 0, stream>>>(
          outp, bufX, bufXh, N, H);
    }
  }
}
